// CausalSelfAttention_53798760350215
// MI455X (gfx1250) — matrix operand order repaired and checked by reading, not compiled, not run on hardware
//
#include <hip/hip_runtime.h>

// ---------------------------------------------------------------------------
// CausalSelfAttention for MI455X (gfx1250, wave32, WMMA bf16)
//   x[B,T,HID] fp32 -> bf16; Wq/Wk/Wv/Wo fp32 -> bf16
//   QKV GEMMs (v_wmma_f32_16x16x32_bf16) + fused RoPE + RMSNorm epilogue
//   Flash attention (online softmax, bf16 WMMA, fp32 accum)
//   Output projection GEMM -> fp32
//   Contiguous LDS staging uses GLOBAL_LOAD_ASYNC_TO_LDS_B128 (ASYNCcnt).
//   Transposed staging keeps all global loads in flight before scattering.
// ---------------------------------------------------------------------------

typedef __bf16 bf16t;
typedef __attribute__((ext_vector_type(16))) __bf16 v16bf;
typedef __attribute__((ext_vector_type(8)))  __bf16 v8bf;
typedef __attribute__((ext_vector_type(8)))  float  v8f;

#define B_   4
#define T_   2048
#define HID_ 2048
#define NH_  16
#define HD_  128

__device__ __forceinline__ bf16t f2bf(float f) {
  unsigned u = __float_as_uint(f);
  u += 0x7FFFu + ((u >> 16) & 1u);          // round-to-nearest-even
  unsigned short h = (unsigned short)(u >> 16);
  bf16t r;
  __builtin_memcpy(&r, &h, 2);
  return r;
}

__device__ __forceinline__ v8f zero8() {
  v8f z;
#pragma unroll
  for (int j = 0; j < 8; ++j) z[j] = 0.f;
  return z;
}

__device__ __forceinline__ v16bf mk16(v8bf lo, v8bf hi) {
  v16bf r;
#pragma unroll
  for (int j = 0; j < 8; ++j) { r[j] = lo[j]; r[j + 8] = hi[j]; }
  return r;
}

// Async copy of one 16-byte chunk per lane: global -> LDS, tracked by ASYNCcnt.
// LDS byte offset = low 32 bits of the generic (flat) shared-aperture address.
__device__ __forceinline__ void async_copy_b128(const bf16t* g, bf16t* lds) {
  unsigned ldsoff = (unsigned)(unsigned long long)(uintptr_t)lds;
  unsigned long long ga = (unsigned long long)(uintptr_t)g;
  asm volatile("global_load_async_to_lds_b128 %0, %1, off"
               :: "v"(ldsoff), "v"(ga) : "memory");
}
__device__ __forceinline__ void wait_async0() {
  asm volatile("s_wait_asynccnt 0x0" ::: "memory");
}

// A fragment (16x32 bf16, row-major source with `stride` elems per row).
// ISA layout: lanes 0-15 (M=lane): elems 0-7 -> K 0..7, elems 8-15 -> K 16..23
//             lanes 16-31: K offset by 8.
__device__ __forceinline__ v16bf load_a_frag(const bf16t* p, int stride,
                                             int mbase, int k0, int lane) {
  int half = lane >> 4, mr = lane & 15;
  const bf16t* row = p + (size_t)(mbase + mr) * stride + k0;
  v8bf lo = *(const v8bf*)(row + half * 8);
  v8bf hi = *(const v8bf*)(row + 16 + half * 8);
  return mk16(lo, hi);
}

// B fragment (32x16 bf16) from an N-major staging buffer: buf[n][k], stride elems
// per n-row. ISA layout: lane = column n (mod 16), elems 0-7 -> K half*8 + j,
// elems 8-15 -> K 16 + half*8 + (j-8).
__device__ __forceinline__ v16bf load_b_frag(const bf16t* p, int stride,
                                             int nbase, int k0, int lane) {
  int half = lane >> 4, n = lane & 15;
  const bf16t* row = p + (size_t)(nbase + n) * stride + k0;
  v8bf lo = *(const v8bf*)(row + half * 8);
  v8bf hi = *(const v8bf*)(row + 16 + half * 8);
  return mk16(lo, hi);
}

// ---------------------------------------------------------------------------
// fp32 -> bf16 conversion
// ---------------------------------------------------------------------------
__global__ void cvt_f32_bf16(const float* __restrict__ in, bf16t* __restrict__ out, int n) {
  int i = blockIdx.x * blockDim.x + threadIdx.x;
  int stride = gridDim.x * blockDim.x;
  for (; i < n; i += stride) out[i] = f2bf(in[i]);
}

// ---------------------------------------------------------------------------
// QKV GEMM + RoPE + RMSNorm.  grid = (NH, B*T/128, 3); block = 256 (8 waves)
// C tile = 128 tokens x 128 dims = one head's full dim for 128 tokens.
// Outputs q/k/v in [B, NH, T, HD] bf16 layout.
// ---------------------------------------------------------------------------
__global__ __launch_bounds__(256) void qkv_kernel(
    const bf16t* __restrict__ xb,
    const bf16t* __restrict__ Wq, const bf16t* __restrict__ Wk, const bf16t* __restrict__ Wv,
    const float* __restrict__ qw, const float* __restrict__ kw,
    bf16t* __restrict__ qo, bf16t* __restrict__ ko, bf16t* __restrict__ vo) {
  __shared__ __align__(16) unsigned char smem[64 * 1024];
  bf16t* As  = (bf16t*)smem;               // [128][32]
  bf16t* Bst = (bf16t*)(smem + 8192);      // [128][32]  (n-major, transposed)
  float* Cs  = (float*)smem;               // [128][128] reused after main loop

  const int tid = threadIdx.x;
  const int lane = tid & 31, wid = tid >> 5;
  const int wm = wid >> 1, wn = wid & 1;   // 4 (M) x 2 (N) wave grid
  const int z = blockIdx.z;
  const bf16t* W = (z == 0) ? Wq : ((z == 1) ? Wk : Wv);
  const int nb = blockIdx.x * 128;         // head h = blockIdx.x
  const int mb = blockIdx.y * 128;

  v8f acc[2][4];
#pragma unroll
  for (int i = 0; i < 2; ++i)
#pragma unroll
    for (int j = 0; j < 4; ++j) acc[i][j] = zero8();

  for (int k0 = 0; k0 < HID_; k0 += 32) {
    // stage A: 128x32 bf16, async global->LDS (2 x b128 per thread)
#pragma unroll
    for (int it = 0; it < 2; ++it) {
      int c = tid + it * 256;
      int r = c >> 2, c8 = (c & 3) * 8;
      async_copy_b128(xb + (size_t)(mb + r) * HID_ + k0 + c8, As + r * 32 + c8);
    }
    // stage B transposed: Bst[n][k].  Issue ALL global loads first so they
    // pipeline, then scatter at 2B granularity.
    v8bf wv_[2];
#pragma unroll
    for (int it = 0; it < 2; ++it) {
      int c = tid + it * 256;
      int kr = c >> 4, n8 = (c & 15) * 8;
      wv_[it] = *(const v8bf*)(W + (size_t)(k0 + kr) * HID_ + nb + n8);
    }
#pragma unroll
    for (int it = 0; it < 2; ++it) {
      int c = tid + it * 256;
      int kr = c >> 4, n8 = (c & 15) * 8;
#pragma unroll
      for (int j = 0; j < 8; ++j) Bst[(n8 + j) * 32 + kr] = wv_[it][j];
    }
    wait_async0();
    __syncthreads();
    if (k0 + 32 < HID_) {  // global_prefetch_b8 next tiles
      __builtin_prefetch(xb + (size_t)(mb + (tid >> 2)) * HID_ + k0 + 32, 0, 1);
      __builtin_prefetch(W + (size_t)(k0 + 32 + (tid >> 4)) * HID_ + nb, 0, 1);
    }
    v16bf bfr[4];
#pragma unroll
    for (int sn = 0; sn < 4; ++sn)
      bfr[sn] = load_b_frag(Bst, 32, wn * 64 + sn * 16, 0, lane);
#pragma unroll
    for (int sm = 0; sm < 2; ++sm) {
      v16bf a = load_a_frag(As, 32, wm * 32 + sm * 16, 0, lane);
#pragma unroll
      for (int sn = 0; sn < 4; ++sn)
        acc[sm][sn] = __builtin_amdgcn_wmma_f32_16x16x32_bf16(
            false, a, false, bfr[sn], (short)0, acc[sm][sn], false, false);
    }
    __syncthreads();
  }

  // spill C tile to LDS (aliases As/Bst; last loop barrier protects this)
  const int half = lane >> 4, ln = lane & 15;
#pragma unroll
  for (int sm = 0; sm < 2; ++sm)
#pragma unroll
    for (int sn = 0; sn < 4; ++sn)
#pragma unroll
      for (int i = 0; i < 8; ++i) {
        int r = wm * 32 + sm * 16 + half * 8 + i;
        int cc = wn * 64 + sn * 16 + ln;
        Cs[r * 128 + cc] = acc[sm][sn][i];
      }
  __syncthreads();

  const int h = blockIdx.x;
  if (z == 2) {
    // V: plain copy to [B,NH,T,HD] bf16
    int r = tid >> 1, d0 = (tid & 1) * 64;
    int row = mb + r, bb = row / T_, t = row % T_;
    bf16t* dst = vo + (((size_t)bb * NH_ + h) * T_ + t) * HD_ + d0;
    const float* cr = Cs + r * 128 + d0;
#pragma unroll 8
    for (int j = 0; j < 64; ++j) dst[j] = f2bf(cr[j]);
  } else if (tid < 128) {
    // Q/K: RoPE then RMSNorm, one token-row per thread
    int r = tid;
    int row = mb + r, bb = row / T_, t = row % T_;
    float* cr = Cs + r * 128;
#pragma unroll 4
    for (int j = 0; j < 64; ++j) {
      // inv_freq = 10000^(-2j/128) = exp(-(2j/128)*ln(10000))
      float inv = __expf(-(float)(2 * j) * (9.2103403719761836f / 128.f));
      float ang = (float)t * inv;
      float cc = __cosf(ang), ss = __sinf(ang);
      float x1 = cr[j], x2 = cr[j + 64];
      cr[j]      = x1 * cc - x2 * ss;
      cr[j + 64] = x2 * cc + x1 * ss;
    }
    float s2 = 0.f;
#pragma unroll 8
    for (int j = 0; j < 128; ++j) s2 += cr[j] * cr[j];
    float rs = rsqrtf(s2 * (1.f / 128.f) + 1e-5f);
    const float* wn2 = (z == 0) ? qw : kw;
    bf16t* dst = ((z == 0) ? qo : ko) + (((size_t)bb * NH_ + h) * T_ + t) * HD_;
#pragma unroll 8
    for (int j = 0; j < 128; ++j) dst[j] = f2bf(cr[j] * rs * wn2[j]);
  }
}

// ---------------------------------------------------------------------------
// Flash attention.  grid = (T/128, B*NH); block = 256 (8 waves).
// Each wave owns 16 query rows; online softmax stats live in registers
// (C-layout replicates each row across a 16-lane half -> shfl_xor reduce).
// ---------------------------------------------------------------------------
__global__ __launch_bounds__(256) void attn_kernel(
    const bf16t* __restrict__ q, const bf16t* __restrict__ k,
    const bf16t* __restrict__ v, bf16t* __restrict__ y) {
  __shared__ __align__(16) bf16t Qs[128 * 128];   // 32 KB
  __shared__ __align__(16) bf16t Ks[64 * 128];    // 16 KB (kv-major)
  __shared__ __align__(16) bf16t Vst[128 * 64];   // 16 KB (d-major, transposed)
  __shared__ __align__(16) bf16t Ps[128 * 64];    // 16 KB (P tile, wave-private rows)

  const int tid = threadIdx.x, lane = tid & 31, wid = tid >> 5;
  const int half = lane >> 4, ln = lane & 15;
  const int qb = blockIdx.x * 128;
  const int bh = blockIdx.y;
  const int bb = bh / NH_, h = bh % NH_;
  const bf16t* qp = q + (size_t)bh * T_ * HD_;
  const bf16t* kp = k + (size_t)bh * T_ * HD_;
  const bf16t* vp = v + (size_t)bh * T_ * HD_;

  // load Q tile (128x128 bf16) async; completion covered by first in-loop wait
#pragma unroll
  for (int it = 0; it < 8; ++it) {
    int c = tid + it * 256;
    int r = c >> 4, d8 = (c & 15) * 8;
    async_copy_b128(qp + (size_t)(qb + r) * HD_ + d8, Qs + r * 128 + d8);
  }

  v8f o[8];
  float rmax[8], rsum[8];
#pragma unroll
  for (int i = 0; i < 8; ++i) { o[i] = zero8(); rmax[i] = -3e38f; rsum[i] = 0.f; }
  const float scale = 0.08838834764831845f;  // 1/sqrt(128)

  for (int kb = 0; kb < qb + 128; kb += 64) {
    __syncthreads();  // protects K/V restaging against previous iteration reads
    // stage K block async (kv-major: rows contiguous in d)
#pragma unroll
    for (int it = 0; it < 4; ++it) {
      int c = tid + it * 256;
      int r = c >> 4, d8 = (c & 15) * 8;
      async_copy_b128(kp + (size_t)(kb + r) * HD_ + d8, Ks + r * 128 + d8);
    }
    // stage V block transposed: Vst[d][kv].  All loads in flight, then scatter.
    v8bf vv_[4];
#pragma unroll
    for (int it = 0; it < 4; ++it) {
      int c = tid + it * 256;
      int r = c >> 4, d8 = (c & 15) * 8;
      vv_[it] = *(const v8bf*)(vp + (size_t)(kb + r) * HD_ + d8);
    }
#pragma unroll
    for (int it = 0; it < 4; ++it) {
      int c = tid + it * 256;
      int r = c >> 4, d8 = (c & 15) * 8;
#pragma unroll
      for (int j = 0; j < 8; ++j) Vst[(d8 + j) * 64 + r] = vv_[it][j];
    }
    wait_async0();     // covers this K block (and the Q tile on iteration 0)
    __syncthreads();

    // S = Q K^T  (16 q-rows x 64 kv-cols per wave, K-dim = HD = 4 x 32)
    v8f s[4];
#pragma unroll
    for (int sn = 0; sn < 4; ++sn) s[sn] = zero8();
#pragma unroll
    for (int ks = 0; ks < 4; ++ks) {
      v16bf a = load_a_frag(Qs, 128, wid * 16, ks * 32, lane);
#pragma unroll
      for (int sn = 0; sn < 4; ++sn) {
        v16bf bb2 = load_b_frag(Ks, 128, sn * 16, ks * 32, lane);
        s[sn] = __builtin_amdgcn_wmma_f32_16x16x32_bf16(
            false, a, false, bb2, (short)0, s[sn], false, false);
      }
    }

    // scale + causal mask + block row-max
    float bm[8];
#pragma unroll
    for (int i = 0; i < 8; ++i) bm[i] = -3e38f;
#pragma unroll
    for (int sn = 0; sn < 4; ++sn) {
      int col = kb + sn * 16 + ln;
#pragma unroll
      for (int i = 0; i < 8; ++i) {
        int rowg = qb + wid * 16 + half * 8 + i;
        float val = s[sn][i] * scale;
        if (col > rowg) val = -3e38f;
        s[sn][i] = val;
        bm[i] = fmaxf(bm[i], val);
      }
    }
#pragma unroll
    for (int i = 0; i < 8; ++i) {  // reduce over the 16-lane half
      float m = bm[i];
      m = fmaxf(m, __shfl_xor(m, 1, 32));
      m = fmaxf(m, __shfl_xor(m, 2, 32));
      m = fmaxf(m, __shfl_xor(m, 4, 32));
      m = fmaxf(m, __shfl_xor(m, 8, 32));
      bm[i] = m;
    }

    // online softmax update
#pragma unroll
    for (int i = 0; i < 8; ++i) {
      float nm = fmaxf(rmax[i], bm[i]);
      float alpha = __expf(rmax[i] - nm);
      rmax[i] = nm;
      float ps = 0.f;
#pragma unroll
      for (int sn = 0; sn < 4; ++sn) {
        float p = __expf(s[sn][i] - nm);
        s[sn][i] = p;
        ps += p;
      }
      ps += __shfl_xor(ps, 1, 32);
      ps += __shfl_xor(ps, 2, 32);
      ps += __shfl_xor(ps, 4, 32);
      ps += __shfl_xor(ps, 8, 32);
      rsum[i] = rsum[i] * alpha + ps;
#pragma unroll
      for (int nt = 0; nt < 8; ++nt) o[nt][i] *= alpha;
    }

    // write P (bf16) — rows are wave-private, same-wave LDS ordering suffices
#pragma unroll
    for (int sn = 0; sn < 4; ++sn)
#pragma unroll
      for (int i = 0; i < 8; ++i)
        Ps[(wid * 16 + half * 8 + i) * 64 + sn * 16 + ln] = f2bf(s[sn][i]);

    // O += P V   (K-dim = 64 kv = 2 x 32)
#pragma unroll
    for (int ks = 0; ks < 2; ++ks) {
      v16bf a = load_a_frag(Ps, 64, wid * 16, ks * 32, lane);
#pragma unroll
      for (int nt = 0; nt < 8; ++nt) {
        v16bf bb2 = load_b_frag(Vst, 64, nt * 16, ks * 32, lane);
        o[nt] = __builtin_amdgcn_wmma_f32_16x16x32_bf16(
            false, a, false, bb2, (short)0, o[nt], false, false);
      }
    }
  }

  // normalize and write y in [B, T, NH*HD] bf16 (row-major for out-proj GEMM)
#pragma unroll
  for (int i = 0; i < 8; ++i) {
    float inv = 1.f / rsum[i];
    int rowg = qb + wid * 16 + half * 8 + i;
    size_t base = ((size_t)bb * T_ + rowg) * (size_t)(NH_ * HD_) + (size_t)h * HD_;
#pragma unroll
    for (int nt = 0; nt < 8; ++nt)
      y[base + nt * 16 + ln] = f2bf(o[nt][i] * inv);
  }
}

// ---------------------------------------------------------------------------
// Output projection: out[B*T,2048] = y_bf16 @ Wo_bf16, fp32 out.
// grid = (HID/128, B*T/128); block = 256.
// ---------------------------------------------------------------------------
__global__ __launch_bounds__(256) void gemm_out_kernel(
    const bf16t* __restrict__ A, const bf16t* __restrict__ W, float* __restrict__ out) {
  __shared__ __align__(16) bf16t As[128 * 32];
  __shared__ __align__(16) bf16t Bst[128 * 32];

  const int tid = threadIdx.x;
  const int lane = tid & 31, wid = tid >> 5;
  const int wm = wid >> 1, wn = wid & 1;
  const int nb = blockIdx.x * 128;
  const int mb = blockIdx.y * 128;

  v8f acc[2][4];
#pragma unroll
  for (int i = 0; i < 2; ++i)
#pragma unroll
    for (int j = 0; j < 4; ++j) acc[i][j] = zero8();

  for (int k0 = 0; k0 < HID_; k0 += 32) {
#pragma unroll
    for (int it = 0; it < 2; ++it) {
      int c = tid + it * 256;
      int r = c >> 2, c8 = (c & 3) * 8;
      async_copy_b128(A + (size_t)(mb + r) * HID_ + k0 + c8, As + r * 32 + c8);
    }
    v8bf wv_[2];
#pragma unroll
    for (int it = 0; it < 2; ++it) {
      int c = tid + it * 256;
      int kr = c >> 4, n8 = (c & 15) * 8;
      wv_[it] = *(const v8bf*)(W + (size_t)(k0 + kr) * HID_ + nb + n8);
    }
#pragma unroll
    for (int it = 0; it < 2; ++it) {
      int c = tid + it * 256;
      int kr = c >> 4, n8 = (c & 15) * 8;
#pragma unroll
      for (int j = 0; j < 8; ++j) Bst[(n8 + j) * 32 + kr] = wv_[it][j];
    }
    wait_async0();
    __syncthreads();
    if (k0 + 32 < HID_) {
      __builtin_prefetch(A + (size_t)(mb + (tid >> 2)) * HID_ + k0 + 32, 0, 1);
      __builtin_prefetch(W + (size_t)(k0 + 32 + (tid >> 4)) * HID_ + nb, 0, 1);
    }
    v16bf bfr[4];
#pragma unroll
    for (int sn = 0; sn < 4; ++sn)
      bfr[sn] = load_b_frag(Bst, 32, wn * 64 + sn * 16, 0, lane);
#pragma unroll
    for (int sm = 0; sm < 2; ++sm) {
      v16bf a = load_a_frag(As, 32, wm * 32 + sm * 16, 0, lane);
#pragma unroll
      for (int sn = 0; sn < 4; ++sn)
        acc[sm][sn] = __builtin_amdgcn_wmma_f32_16x16x32_bf16(
            false, a, false, bfr[sn], (short)0, acc[sm][sn], false, false);
    }
    __syncthreads();
  }

  const int half = lane >> 4, ln = lane & 15;
#pragma unroll
  for (int sm = 0; sm < 2; ++sm)
#pragma unroll
    for (int sn = 0; sn < 4; ++sn)
#pragma unroll
      for (int i = 0; i < 8; ++i) {
        int r = mb + wm * 32 + sm * 16 + half * 8 + i;
        int cc = nb + wn * 64 + sn * 16 + ln;
        out[(size_t)r * HID_ + cc] = acc[sm][sn][i];
      }
}

// ---------------------------------------------------------------------------
// Host launcher
// ---------------------------------------------------------------------------
extern "C" void kernel_launch(void* const* d_in, const int* in_sizes, int n_in,
                              void* d_out, int out_size, void* d_ws, size_t ws_size,
                              hipStream_t stream) {
  const float* x   = (const float*)d_in[0];
  const float* Wqf = (const float*)d_in[1];
  const float* Wkf = (const float*)d_in[2];
  const float* Wvf = (const float*)d_in[3];
  const float* Wof = (const float*)d_in[4];
  const float* qw  = (const float*)d_in[5];
  const float* kw  = (const float*)d_in[6];

  unsigned char* ws = (unsigned char*)d_ws;
  size_t off = 0;
  auto alloc = [&](size_t bytes) -> void* {
    void* p = ws + off;
    off = (off + bytes + 255) & ~(size_t)255;
    return p;
  };
  const size_t xN = (size_t)B_ * T_ * HID_;          // 16.8M elems
  const size_t wN = (size_t)HID_ * NH_ * HD_;        // 4.2M elems
  bf16t* xb  = (bf16t*)alloc(xN * 2);
  bf16t* wqb = (bf16t*)alloc(wN * 2);
  bf16t* wkb = (bf16t*)alloc(wN * 2);
  bf16t* wvb = (bf16t*)alloc(wN * 2);
  bf16t* wob = (bf16t*)alloc(wN * 2);
  bf16t* qb  = (bf16t*)alloc(xN * 2);
  bf16t* kb  = (bf16t*)alloc(xN * 2);
  bf16t* vb  = (bf16t*)alloc(xN * 2);
  bf16t* yb  = (bf16t*)alloc(xN * 2);

  cvt_f32_bf16<<<4096, 256, 0, stream>>>(x,   xb,  (int)xN);
  cvt_f32_bf16<<<2048, 256, 0, stream>>>(Wqf, wqb, (int)wN);
  cvt_f32_bf16<<<2048, 256, 0, stream>>>(Wkf, wkb, (int)wN);
  cvt_f32_bf16<<<2048, 256, 0, stream>>>(Wvf, wvb, (int)wN);
  cvt_f32_bf16<<<2048, 256, 0, stream>>>(Wof, wob, (int)wN);

  dim3 g1(NH_, (B_ * T_) / 128, 3);
  qkv_kernel<<<g1, 256, 0, stream>>>(xb, wqb, wkb, wvb, qw, kw, qb, kb, vb);

  dim3 g2(T_ / 128, B_ * NH_);
  attn_kernel<<<g2, 256, 0, stream>>>(qb, kb, vb, yb);

  dim3 g3(HID_ / 128, (B_ * T_) / 128);
  gemm_out_kernel<<<g3, 256, 0, stream>>>(yb, wob, (float*)d_out);
}
